// HyenaFilter_6725918785553
// MI455X (gfx1250) — matrix operand order repaired and checked by reading, not compiled, not run on hardware
//
#include <hip/hip_runtime.h>
#include <hip/hip_bf16.h>
#include <math.h>

// ---------------------------------------------------------------------------
// Hyena long-conv for MI455X (gfx1250, wave32, WMMA).
//   y[b,t,c] = sum_{s<=t} u[b,s,c] * k[t-s,c]  +  u[b,t,c]*bias[c]
// Causal conv == lower-triangular Toeplitz matmul per channel, N = batch(16)
// -> v_wmma_f32_16x16x32_f16 tiles, f32 accumulation.
//  K1: u[B,L,CH] f32 -> uT[c][b][s] f16      (L2-resident, 101 MB)
//  K2: implicit filter MLP -> reversed padded f16 filter krev (6 MB)
//  K3: WMMA Toeplitz conv  -> yT[c][b][t] f32 (contiguous 32B stores)
//      - filter staged to LDS via async global->LDS DMA (ASYNCcnt path)
//      - 8 shift-copies in LDS so A fragments are aligned ds b128 loads
//  K4: LDS-tiled transpose yT -> y[b,t,c] + fused u*bias (fully coalesced)
// ---------------------------------------------------------------------------

typedef __attribute__((ext_vector_type(16))) _Float16 v16h;
typedef __attribute__((ext_vector_type(8)))  _Float16 v8h;
typedef __attribute__((ext_vector_type(8)))  float    v8f;

#define L_SEQ   4096
#define CH      768
#define NB      16            // batch
#define FO      16            // filter order
#define LP      (L_SEQ + 32)  // padded row length of transposed u (=4128)
#define PREV    4144          // reversed+padded filter length per channel

// workspace layout (total ~309 MB)
#define UT_BYTES   ((size_t)CH * NB * LP * 2)        // 101,449,728
#define KREV_OFF   UT_BYTES
#define KREV_BYTES ((size_t)CH * PREV * 2)           //   6,365,184
#define YT_OFF     (KREV_OFF + KREV_BYTES)
#define YT_BYTES   ((size_t)CH * NB * L_SEQ * 4)     // 201,326,592

// ---------------------------------------------------------------------------
// Kernel 1: u[B,L,CH] f32 -> uT[c][b][s] f16 (row stride LP, tail zeroed).
// ---------------------------------------------------------------------------
__global__ void hyena_transpose_cast(const float* __restrict__ uin,
                                     _Float16* __restrict__ uT) {
    const int c  = blockIdx.x * 256 + threadIdx.x;
    const int s0 = blockIdx.y * 8;
    const int b  = blockIdx.z;
    v8h r;
#pragma unroll
    for (int i = 0; i < 8; ++i)
        r[i] = (_Float16)uin[((size_t)b * L_SEQ + (s0 + i)) * CH + c];
    _Float16* row = uT + (size_t)(c * NB + b) * LP;
    *(v8h*)(row + s0) = r;
    if (s0 == 0) {             // zero the 32-tap overrun tail once per row
        v8h z = {};
#pragma unroll
        for (int j = 0; j < 4; ++j) *(v8h*)(row + L_SEQ + j * 8) = z;
    }
}

// ---------------------------------------------------------------------------
// Kernel 2: implicit filter MLP -> reversed, zero-padded fp16 filter.
//   kpad[p] = (32 <= p < L+32) ? k[p-32][c] : 0 ;  krev[q] = kpad[PREV-1-q]
// ---------------------------------------------------------------------------
__global__ void hyena_filter(const float* __restrict__ w1, const float* __restrict__ b1,
                             const float* __restrict__ w2, const float* __restrict__ b2,
                             const float* __restrict__ w3, const float* __restrict__ b3,
                             const float* __restrict__ freq,
                             const float* __restrict__ deltas,
                             _Float16* __restrict__ krev) {
    const int pos = blockIdx.x;
    const int tid = threadIdx.x;

    const float tn   = (float)pos / (float)(L_SEQ - 1);
    const float wang = 2.0f * 3.14159265358979323846f * (float)pos / (float)L_SEQ;
    const float z0 = tn, z1 = cosf(1e-4f * wang), z2 = -sinf(1e-4f * wang);

    float h1[FO], h2[FO];
#pragma unroll
    for (int o = 0; o < FO; ++o) {
        float s = w1[o * 3 + 0] * z0 + w1[o * 3 + 1] * z1 + w1[o * 3 + 2] * z2 + b1[o];
        h1[o] = sinf(freq[o] * s);
    }
#pragma unroll
    for (int o = 0; o < FO; ++o) {
        float s = b2[o];
#pragma unroll
        for (int i = 0; i < FO; ++i) s += w2[o * FO + i] * h1[i];
        h2[o] = sinf(freq[o] * s);
    }
    for (int c = tid; c < CH; c += 256) {
        float s = b3[c];
#pragma unroll
        for (int i = 0; i < FO; ++i) s += w3[c * FO + i] * h2[i];
        s *= expf(-tn * deltas[c]);
        krev[(size_t)c * PREV + (PREV - 33 - pos)] = (_Float16)s;
    }
    if (pos == 0) {   // causal / tail zero pads
        for (int idx = tid; idx < CH * 48; idx += 256) {
            int c = idx / 48, j = idx % 48;
            int q = (j < 16) ? j : (PREV - 32 + (j - 16));
            krev[(size_t)c * PREV + q] = (_Float16)0.0f;
        }
    }
}

// ---------------------------------------------------------------------------
// Kernel 3: WMMA Toeplitz convolution. Block = 8 waves, one channel.
// Wave w: output rows t in [T0,T0+16), T0 = 128*blockIdx.y + 16*w.
// ---------------------------------------------------------------------------
__global__ void __launch_bounds__(256)
hyena_conv_wmma(const _Float16* __restrict__ uT,
                const _Float16* __restrict__ krev,
                float* __restrict__ yT) {
    // 8 shift-copies: klds8[r][i] = krev[c][i+r]  (A runs become 16B-aligned)
    __shared__ __align__(16) _Float16 klds8[8][PREV];

    const int c    = blockIdx.x;
    const int tid  = threadIdx.x;
    const int wave = tid >> 5;
    const int lane = tid & 31;
    const int grp  = lane >> 4;
    const int ln16 = lane & 15;

    // ---- stage copy 0 via async global->LDS DMA (ASYNCcnt path) ----
    {
        const unsigned long long gbase =
            (unsigned long long)(uintptr_t)(krev + (size_t)c * PREV);
        const unsigned lbase = (unsigned)(uintptr_t)&klds8[0][0]; // LDS offset
        for (int i = tid; i < (PREV * 2) / 16; i += 256) {
            unsigned loff = lbase + (unsigned)i * 16u;
            unsigned goff = (unsigned)i * 16u;
            asm volatile("global_load_async_to_lds_b128 %0, %1, %2"
                         :: "v"(loff), "v"(goff), "s"(gbase) : "memory");
        }
        asm volatile("s_wait_asynccnt 0" ::: "memory");
    }
    __syncthreads();
    // ---- derive shifted copies 1..7 ----
#pragma unroll
    for (int r = 1; r < 8; ++r)
        for (int i = tid; i < PREV - r; i += 256)
            klds8[r][i] = klds8[0][i + r];
    __syncthreads();

    const int T0 = blockIdx.y * 128 + wave * 16;
    const _Float16* urow = uT + (size_t)(c * NB + ln16) * LP;

    // One K=32 chunk: A[M][j]=k[(T0-S0)+M-j]=krev[qbase+j]; B=u(32s x 16b)
    auto chunk = [&](int S0, v8f acc) -> v8f {
        const int qbase = PREV - 33 - (T0 - S0) - ln16;
        const int S1 = qbase + grp * 8;      // start of K=grp*8+e run
        const int r  = S1 & 7;
        const _Float16* ab = &klds8[r][S1 - r];          // 16B aligned
        v8h alo = *(const v8h*)(ab);                      // K = grp*8   + 0..7
        v8h ahi = *(const v8h*)(ab + 16);                 // K = 16+grp*8+ 0..7
        v16h a = __builtin_shufflevector(alo, ahi,
                 0,1,2,3,4,5,6,7,8,9,10,11,12,13,14,15);
        v8h blo = *(const v8h*)(urow + S0 + grp * 8);        // K = grp*8   + 0..7
        v8h bhi = *(const v8h*)(urow + S0 + 16 + grp * 8);   // K = 16+grp*8+ 0..7
        v16h bm = __builtin_shufflevector(blo, bhi,
                 0,1,2,3,4,5,6,7,8,9,10,11,12,13,14,15);
        return __builtin_amdgcn_wmma_f32_16x16x32_f16(
            false, a, false, bm, (short)0, acc, false, false);
    };

    v8f acc0 = {}, acc1 = {};
    int S0 = 0;
    for (; S0 + 32 <= T0; S0 += 64) {        // 2 chunks, independent accums
        __builtin_prefetch(urow + S0 + 128, 0, 1);
        acc0 = chunk(S0,      acc0);
        acc1 = chunk(S0 + 32, acc1);
    }
    if (S0 <= T0) acc0 = chunk(S0, acc0);    // remainder chunk
    v8f acc = acc0 + acc1;

    // D lane: N=ln16 (=batch), VGPR r -> t = T0 + 8*grp + r  (contiguous!)
    *(v8f*)(yT + ((size_t)(c * NB + ln16)) * L_SEQ + T0 + 8 * grp) = acc;
}

// ---------------------------------------------------------------------------
// Kernel 4: yT[c][b][t] -> y[b][t][c] (LDS 32x32 tile transpose) + u*bias.
// ---------------------------------------------------------------------------
__global__ void hyena_bias_transpose(const float* __restrict__ yT,
                                     const float* __restrict__ uin,
                                     const float* __restrict__ bias,
                                     float* __restrict__ out) {
    __shared__ float tile[32][33];
    const int c0 = blockIdx.x * 32, t0 = blockIdx.y * 32, b = blockIdx.z;
    const int tt = threadIdx.x & 31, cg = threadIdx.x >> 5;
#pragma unroll
    for (int k = 0; k < 4; ++k) {
        const int cc = cg + 8 * k;
        tile[cc][tt] = yT[((size_t)((c0 + cc) * NB + b)) * L_SEQ + t0 + tt];
    }
    __syncthreads();
    const int co = threadIdx.x & 31, tg = threadIdx.x >> 5;
    const float bc = bias[c0 + co];
#pragma unroll
    for (int k = 0; k < 4; ++k) {
        const int t = t0 + tg + 8 * k;
        const size_t off = ((size_t)b * L_SEQ + t) * CH + c0 + co;
        out[off] = tile[co][tg + 8 * k] + uin[off] * bc;
    }
}

// ---------------------------------------------------------------------------
extern "C" void kernel_launch(void* const* d_in, const int* in_sizes, int n_in,
                              void* d_out, int out_size, void* d_ws, size_t ws_size,
                              hipStream_t stream) {
    const float* u      = (const float*)d_in[0];
    const float* w1     = (const float*)d_in[1];
    const float* b1     = (const float*)d_in[2];
    const float* w2     = (const float*)d_in[3];
    const float* b2     = (const float*)d_in[4];
    const float* w3     = (const float*)d_in[5];
    const float* b3     = (const float*)d_in[6];
    const float* freq   = (const float*)d_in[7];
    const float* deltas = (const float*)d_in[8];
    const float* bias   = (const float*)d_in[9];
    float* out = (float*)d_out;

    _Float16* uT   = (_Float16*)d_ws;
    _Float16* krev = (_Float16*)((char*)d_ws + KREV_OFF);
    float*    yT   = (float*)((char*)d_ws + YT_OFF);
    (void)in_sizes; (void)n_in; (void)out_size; (void)ws_size;

    hipLaunchKernelGGL(hyena_transpose_cast, dim3(CH / 256, L_SEQ / 8, NB),
                       dim3(256), 0, stream, u, uT);
    hipLaunchKernelGGL(hyena_filter, dim3(L_SEQ), dim3(256), 0, stream,
                       w1, b1, w2, b2, w3, b3, freq, deltas, krev);
    hipLaunchKernelGGL(hyena_conv_wmma, dim3(CH, L_SEQ / 128), dim3(256), 0, stream,
                       uT, krev, yT);
    hipLaunchKernelGGL(hyena_bias_transpose, dim3(CH / 32, L_SEQ / 32, NB),
                       dim3(256), 0, stream, yT, u, bias, out);
}
